// YOCOGatedRetention_22909355556915
// MI455X (gfx1250) — matrix operand order repaired and checked by reading, not compiled, not run on hardware
//
#include <hip/hip_runtime.h>
#include <hip/hip_bf16.h>
#include <math.h>

#define HDIM 2048
#define TLEN 2048
#define NH 32
#define DH 64

typedef __bf16 bf16;
typedef float v8f   __attribute__((ext_vector_type(8)));
typedef bf16  v16bf __attribute__((ext_vector_type(16)));
typedef bf16  v8bf  __attribute__((ext_vector_type(8)));

#if __has_builtin(__builtin_amdgcn_global_load_async_to_lds_b128)
#define RET_ASYNC 1
typedef int v4i_g __attribute__((vector_size(16)));           // GCC-vector int4
typedef __attribute__((address_space(1))) v4i_g* gv4i_p;      // global side
typedef __attribute__((address_space(3))) v4i_g* lv4i_p;      // LDS side
#endif

static __device__ __forceinline__ v16bf cat8(v8bf lo, v8bf hi) {
    return __builtin_shufflevector(lo, hi, 0,1,2,3,4,5,6,7,8,9,10,11,12,13,14,15);
}
static __device__ __forceinline__ v8f wmma_bf16(v16bf a, v16bf b, v8f c) {
    return __builtin_amdgcn_wmma_f32_16x16x32_bf16(false, a, false, b, (short)0, c, false, false);
}
static __device__ __forceinline__ unsigned pack_bf16(float x, float y) {
    unsigned short ua = __builtin_bit_cast(unsigned short, (bf16)x);
    unsigned short ub = __builtin_bit_cast(unsigned short, (bf16)y);
    return (unsigned)ua | ((unsigned)ub << 16);
}

// ---------------------------------------------------------------------------
// C[M,N] = A[M,K] * B[N,K]^T   (B fp32 weights; A fp32 or bf16)
// 256 thr (8 waves), tile 128x128, wave -> 32(M) x 64(N), K step 32.
// Register-staged double buffering: one barrier per K step.
// ---------------------------------------------------------------------------
template <typename AT>
__global__ __launch_bounds__(256)
void gemm_bt(const AT* __restrict__ A, const float* __restrict__ B,
             float* __restrict__ C, int M, int N, int K) {
    __shared__ __align__(16) bf16 As[2][128][40];
    __shared__ __align__(16) bf16 Bs[2][128][40];
    const int tid  = threadIdx.x;
    const int lane = tid & 31, w = tid >> 5;
    const int lh = lane >> 4, ln = lane & 15;
    const int m0 = blockIdx.y * 128, n0 = blockIdx.x * 128;
    const int msub = (w & 3) * 32, nsub = (w >> 2) * 64;

    v8f zero = {};
    v8f acc[2][4];
    #pragma unroll
    for (int mi = 0; mi < 2; ++mi)
        #pragma unroll
        for (int ni = 0; ni < 4; ++ni) acc[mi][ni] = zero;

    const int tr = tid >> 1;             // 0..127
    const int tc = (tid & 1) * 16;       // 0 or 16 elements

    float4 brg[4];      // B slab staging (fp32)
    float4 afr4[4];     // A slab staging when AT == float
    uint4  ahr[2];      // A slab staging when AT == bf16

    auto fetch = [&](int k0) {
        size_t aoff = (size_t)(m0 + tr) * K + k0 + tc;
        if constexpr (sizeof(AT) == 4) {
            const float4* p = reinterpret_cast<const float4*>(A + aoff);
            afr4[0] = p[0]; afr4[1] = p[1]; afr4[2] = p[2]; afr4[3] = p[3];
        } else {
            const uint4* p = reinterpret_cast<const uint4*>(A + aoff);
            ahr[0] = p[0]; ahr[1] = p[1];
        }
        size_t boff = (size_t)(n0 + tr) * K + k0 + tc;
        const float4* q = reinterpret_cast<const float4*>(B + boff);
        brg[0] = q[0]; brg[1] = q[1]; brg[2] = q[2]; brg[3] = q[3];
    };
    auto commit = [&](int b) {
        if constexpr (sizeof(AT) == 4) {
            #pragma unroll
            for (int j = 0; j < 4; ++j) {
                int o = tc + j * 4;
                *reinterpret_cast<unsigned*>(&As[b][tr][o])     = pack_bf16(afr4[j].x, afr4[j].y);
                *reinterpret_cast<unsigned*>(&As[b][tr][o + 2]) = pack_bf16(afr4[j].z, afr4[j].w);
            }
        } else {
            *reinterpret_cast<uint4*>(&As[b][tr][tc])     = ahr[0];
            *reinterpret_cast<uint4*>(&As[b][tr][tc + 8]) = ahr[1];
        }
        #pragma unroll
        for (int j = 0; j < 4; ++j) {
            int o = tc + j * 4;
            *reinterpret_cast<unsigned*>(&Bs[b][tr][o])     = pack_bf16(brg[j].x, brg[j].y);
            *reinterpret_cast<unsigned*>(&Bs[b][tr][o + 2]) = pack_bf16(brg[j].z, brg[j].w);
        }
    };

    fetch(0);
    commit(0);
    __syncthreads();

    int buf = 0;
    for (int k0 = 0; k0 < K; k0 += 32) {
        const bool more = (k0 + 32) < K;
        if (more) fetch(k0 + 32);        // HBM/L2 latency overlaps the WMMAs below

        // hoist all fragments, then a clean WMMA chain
        v16bf af[2], bfm[4];
        #pragma unroll
        for (int mi = 0; mi < 2; ++mi) {
            int row = msub + mi * 16 + ln;
            af[mi] = cat8(*reinterpret_cast<const v8bf*>(&As[buf][row][lh * 8]),
                          *reinterpret_cast<const v8bf*>(&As[buf][row][16 + lh * 8]));
        }
        #pragma unroll
        for (int ni = 0; ni < 4; ++ni) {
            int col = nsub + ni * 16 + ln;
            bfm[ni] = cat8(*reinterpret_cast<const v8bf*>(&Bs[buf][col][lh * 8]),
                           *reinterpret_cast<const v8bf*>(&Bs[buf][col][16 + lh * 8]));
        }
        #pragma unroll
        for (int ni = 0; ni < 4; ++ni)
            #pragma unroll
            for (int mi = 0; mi < 2; ++mi)
                acc[mi][ni] = wmma_bf16(af[mi], bfm[ni], acc[mi][ni]);

        if (more) {
            commit(buf ^ 1);
            __syncthreads();
            buf ^= 1;
        }
    }

    #pragma unroll
    for (int mi = 0; mi < 2; ++mi)
        #pragma unroll
        for (int ni = 0; ni < 4; ++ni) {
            int col = n0 + nsub + ni * 16 + ln;
            #pragma unroll
            for (int r = 0; r < 8; ++r) {
                int row = m0 + msub + mi * 16 + r + 8 * lh;
                C[(size_t)row * N + col] = acc[mi][ni][r];
            }
        }
}

// ---------------------------------------------------------------------------
// gk[t,h] = logsigmoid(hidden[t,:] . Wgk[h,:]) / 16    (one wave per output)
// ---------------------------------------------------------------------------
__global__ __launch_bounds__(256)
void gk_kernel(const float* __restrict__ X, const float* __restrict__ Wgk,
               float* __restrict__ gk) {
    int w = threadIdx.x >> 5, lane = threadIdx.x & 31;
    int o = blockIdx.x * 8 + w;
    int t = o >> 5;                  // H == 32
    int h = o & 31;
    const float* x  = X   + (size_t)t * HDIM;
    const float* wv = Wgk + (size_t)h * HDIM;
    float s = 0.f;
    for (int i = lane; i < HDIM; i += 32) s += x[i] * wv[i];
    #pragma unroll
    for (int off = 16; off; off >>= 1) s += __shfl_down(s, off, 32);
    if (lane == 0) {
        float ls = fminf(s, 0.f) - log1pf(__expf(-fabsf(s)));
        gk[o] = ls * (1.0f / 16.0f);
    }
}

// ---------------------------------------------------------------------------
// Inclusive cumsum over T per head: one wave per head, wave-scan in 32-chunks
// ---------------------------------------------------------------------------
__global__ void cumsum_kernel(const float* __restrict__ gk, float* __restrict__ G) {
    int h = blockIdx.x, lane = threadIdx.x;
    float run = 0.f;
    for (int t0 = 0; t0 < TLEN; t0 += 32) {
        float v = gk[(size_t)(t0 + lane) * NH + h];
        #pragma unroll
        for (int off = 1; off < 32; off <<= 1) {
            float u = __shfl_up(v, off, 32);
            if (lane >= off) v += u;
        }
        G[(size_t)(t0 + lane) * NH + h] = run + v;
        run += __shfl(v, 31, 32);
    }
}

// ---------------------------------------------------------------------------
// Interleaved RoPE, fp32 in -> bf16 out. One thread per (even,odd) pair.
// ---------------------------------------------------------------------------
__global__ __launch_bounds__(256)
void rope_kernel(const float* __restrict__ in, bf16* __restrict__ out) {
    int idx = blockIdx.x * blockDim.x + threadIdx.x;
    int t   = idx / (HDIM / 2);
    int rem = idx % (HDIM / 2);                        // rem = 32*h + d2
    int d2  = rem & 31;
    float inv = __expf(-((float)(2 * d2) * (1.0f / 64.0f)) * 9.210340371976184f);
    float fr  = (float)t * inv;
    float s, c;
    __sincosf(fr, &s, &c);
    size_t base = (size_t)t * HDIM + (size_t)rem * 2;
    float x1 = in[base], x2 = in[base + 1];
    out[base]     = (bf16)(x1 * c - x2 * s);
    out[base + 1] = (bf16)(x1 * s + x2 * c);
}

__global__ __launch_bounds__(256)
void cast_kernel(const float* __restrict__ in, bf16* __restrict__ out, int n) {
    int i = blockIdx.x * blockDim.x + threadIdx.x;
    if (i < n) out[i] = (bf16)in[i];
}

// ---------------------------------------------------------------------------
// Gated retention: per (head, 64-row i-tile); 128 thr = 4 waves, wave = 16 rows
// K tile via async global->LDS copy when available; V transposed with packed
// b32 stores; next j-tile prefetched into registers during compute.
// ---------------------------------------------------------------------------
__global__ __launch_bounds__(128)
void attn_kernel(const bf16* __restrict__ qb, const bf16* __restrict__ kbuf,
                 const bf16* __restrict__ vbuf, const float* __restrict__ G,
                 const float* __restrict__ gf, bf16* __restrict__ ob) {
    __shared__ __align__(16) bf16 Qs[64][72];
    __shared__ __align__(16) bf16 Ks[64][72];
    __shared__ __align__(16) bf16 Vts[64][72];      // V transposed: [d][j]
    __shared__ __align__(16) bf16 Aps[4][16][72];   // per-wave prob tile
    __shared__ float SGi[64], SGj[64];

    const int tid  = threadIdx.x;
    const int lane = tid & 31, w = tid >> 5;
    const int lh = lane >> 4, ln = lane & 15;
    const int h  = blockIdx.x;
    const int i0 = blockIdx.y * 64;
    const float scale = 0.125f;   // D^-0.5

    const int lr  = tid >> 1, lcb = (tid & 1) * 32;   // K/Q loader mapping
    const int vj  = (tid & 31) * 2;                   // V loader: rows vj, vj+1
    const int vdb = (tid >> 5) * 16;                  // V loader: 16 d columns

    // ---- load Q tile + G_i ----
    {
        const uint4* qp = reinterpret_cast<const uint4*>(
            qb + (size_t)(i0 + lr) * HDIM + h * 64 + lcb);
        #pragma unroll
        for (int j = 0; j < 4; ++j)
            *reinterpret_cast<uint4*>(&Qs[lr][lcb + j * 8]) = qp[j];
        if (tid < 64) SGi[tid] = G[(size_t)(i0 + tid) * NH + h];
    }
    __syncthreads();

    float gi[8];
    #pragma unroll
    for (int r = 0; r < 8; ++r) gi[r] = SGi[w * 16 + r + 8 * lh];

    v16bf qa[2];
    #pragma unroll
    for (int kc = 0; kc < 2; ++kc) {
        int row = w * 16 + ln;
        qa[kc] = cat8(*reinterpret_cast<const v8bf*>(&Qs[row][kc * 32 + lh * 8]),
                      *reinterpret_cast<const v8bf*>(&Qs[row][kc * 32 + 16 + lh * 8]));
    }

    v8f zero = {};
    v8f oacc[4] = {zero, zero, zero, zero};

    // register staging for the next j tile
    union V16 { uint4 u[2]; unsigned short s[16]; };
    V16 vlo, vhi;
    float gjr = 0.f;
#ifndef RET_ASYNC
    uint4 kreg[4];
#endif

    auto fetchKV = [&](int j0) {
#ifndef RET_ASYNC
        const uint4* kp = reinterpret_cast<const uint4*>(
            kbuf + (size_t)(j0 + lr) * HDIM + h * 64 + lcb);
        #pragma unroll
        for (int j = 0; j < 4; ++j) kreg[j] = kp[j];
#endif
        const uint4* p0 = reinterpret_cast<const uint4*>(
            vbuf + (size_t)(j0 + vj) * HDIM + h * 64 + vdb);
        vlo.u[0] = p0[0]; vlo.u[1] = p0[1];
        const uint4* p1 = reinterpret_cast<const uint4*>(
            vbuf + (size_t)(j0 + vj + 1) * HDIM + h * 64 + vdb);
        vhi.u[0] = p1[0]; vhi.u[1] = p1[1];
        gjr = (tid < 64) ? G[(size_t)(j0 + tid) * NH + h] : 0.f;
    };

    fetchKV(0);

    for (int j0 = 0; j0 <= i0; j0 += 64) {
        // ---- commit tile into LDS (Ks/Vts free: first iter or post-barrier) ----
#ifdef RET_ASYNC
        {
            const bf16* gk0 = kbuf + (size_t)(j0 + lr) * HDIM + h * 64 + lcb;
            #pragma unroll
            for (int j = 0; j < 4; ++j)
                __builtin_amdgcn_global_load_async_to_lds_b128(
                    (gv4i_p)(gk0 + j * 8),
                    (lv4i_p)&Ks[lr][lcb + j * 8],
                    0, 0);
        }
#else
        #pragma unroll
        for (int j = 0; j < 4; ++j)
            *reinterpret_cast<uint4*>(&Ks[lr][lcb + j * 8]) = kreg[j];
#endif
        #pragma unroll
        for (int i = 0; i < 16; ++i) {
            unsigned pk = (unsigned)vlo.s[i] | ((unsigned)vhi.s[i] << 16);
            *reinterpret_cast<unsigned*>(&Vts[vdb + i][vj]) = pk;
        }
        if (tid < 64) SGj[tid] = gjr;
#ifdef RET_ASYNC
#if __has_builtin(__builtin_amdgcn_s_wait_asynccnt)
        __builtin_amdgcn_s_wait_asynccnt(0);
#else
        asm volatile("s_wait_asynccnt 0" ::: "memory");
#endif
#endif
        __syncthreads();

        if (j0 + 64 <= i0) fetchKV(j0 + 64);   // overlap next tile with compute

        // ---- S = Q K^T : hoisted fragments, back-to-back WMMAs ----
        v8f sacc[4] = {zero, zero, zero, zero};
        #pragma unroll
        for (int kc = 0; kc < 2; ++kc) {
            v16bf bfr[4];
            #pragma unroll
            for (int nt = 0; nt < 4; ++nt) {
                int jrow = nt * 16 + ln;
                bfr[nt] = cat8(
                    *reinterpret_cast<const v8bf*>(&Ks[jrow][kc * 32 + lh * 8]),
                    *reinterpret_cast<const v8bf*>(&Ks[jrow][kc * 32 + 16 + lh * 8]));
            }
            #pragma unroll
            for (int nt = 0; nt < 4; ++nt)
                sacc[nt] = wmma_bf16(qa[kc], bfr[nt], sacc[nt]);
        }
        // ---- gated causal epilogue, write prob tile ----
        #pragma unroll
        for (int nt = 0; nt < 4; ++nt) {
            float gj = SGj[nt * 16 + ln];
            int   jg = j0 + nt * 16 + ln;
            #pragma unroll
            for (int r = 0; r < 8; ++r) {
                int ig = i0 + w * 16 + r + 8 * lh;
                float val = 0.f;
                if (ig >= jg) val = __expf(gi[r] - gj) * sacc[nt][r] * scale;
                Aps[w][r + 8 * lh][nt * 16 + ln] = (bf16)val;
            }
        }
        __syncthreads();

        // ---- O += A V ----
        #pragma unroll
        for (int jc = 0; jc < 2; ++jc) {
            v16bf afr = cat8(
                *reinterpret_cast<const v8bf*>(&Aps[w][ln][jc * 32 + lh * 8]),
                *reinterpret_cast<const v8bf*>(&Aps[w][ln][jc * 32 + 16 + lh * 8]));
            v16bf bfr[4];
            #pragma unroll
            for (int dt = 0; dt < 4; ++dt) {
                int drow = dt * 16 + ln;
                bfr[dt] = cat8(
                    *reinterpret_cast<const v8bf*>(&Vts[drow][jc * 32 + lh * 8]),
                    *reinterpret_cast<const v8bf*>(&Vts[drow][jc * 32 + 16 + lh * 8]));
            }
            #pragma unroll
            for (int dt = 0; dt < 4; ++dt)
                oacc[dt] = wmma_bf16(afr, bfr[dt], oacc[dt]);
        }
        __syncthreads();
    }

    // ---- RMS norm over D + silu(gate), store bf16 ----
    #pragma unroll
    for (int r = 0; r < 8; ++r) {
        float ss = 0.f;
        #pragma unroll
        for (int dt = 0; dt < 4; ++dt) ss += oacc[dt][r] * oacc[dt][r];
        #pragma unroll
        for (int m = 8; m; m >>= 1) ss += __shfl_xor(ss, m, 32);
        float rs = rsqrtf(ss * (1.0f / 64.0f) + 1e-6f);
        int ig = i0 + w * 16 + r + 8 * lh;
        #pragma unroll
        for (int dt = 0; dt < 4; ++dt) {
            int d = dt * 16 + ln;
            float gate = gf[(size_t)ig * HDIM + h * 64 + d];
            float sil  = gate / (1.0f + __expf(-gate));
            ob[(size_t)ig * HDIM + h * 64 + d] = (bf16)(oacc[dt][r] * rs * sil);
        }
    }
}

// ---------------------------------------------------------------------------
extern "C" void kernel_launch(void* const* d_in, const int* in_sizes, int n_in,
                              void* d_out, int out_size, void* d_ws, size_t ws_size,
                              hipStream_t stream) {
    (void)in_sizes; (void)n_in; (void)out_size; (void)ws_size;
    const float* hidden = (const float*)d_in[0];
    const float* Wq  = (const float*)d_in[1];
    const float* Wk  = (const float*)d_in[2];
    const float* Wv  = (const float*)d_in[3];
    const float* Wg  = (const float*)d_in[4];
    const float* Wgk = (const float*)d_in[5];
    const float* Wo  = (const float*)d_in[6];
    float* out = (float*)d_out;

    char* p = (char*)d_ws;
    const size_t fsz = (size_t)TLEN * HDIM * sizeof(float);
    const size_t hsz = (size_t)TLEN * HDIM * sizeof(bf16);
    float* qf = (float*)p; p += fsz;
    float* kf = (float*)p; p += fsz;
    float* vf = (float*)p; p += fsz;
    float* gf = (float*)p; p += fsz;
    bf16* qbm = (bf16*)p;  p += hsz;
    bf16* kbm = (bf16*)p;  p += hsz;
    bf16* vbm = (bf16*)p;  p += hsz;
    bf16* obm = (bf16*)p;  p += hsz;
    float* gkbuf = (float*)p; p += (size_t)TLEN * NH * sizeof(float);
    float* Gbuf  = (float*)p; p += (size_t)TLEN * NH * sizeof(float);

    dim3 gg(HDIM / 128, TLEN / 128);
    gemm_bt<float><<<gg, 256, 0, stream>>>(hidden, Wq, qf, TLEN, HDIM, HDIM);
    gemm_bt<float><<<gg, 256, 0, stream>>>(hidden, Wk, kf, TLEN, HDIM, HDIM);
    gemm_bt<float><<<gg, 256, 0, stream>>>(hidden, Wv, vf, TLEN, HDIM, HDIM);
    gemm_bt<float><<<gg, 256, 0, stream>>>(hidden, Wg, gf, TLEN, HDIM, HDIM);
    gk_kernel<<<TLEN * NH / 8, 256, 0, stream>>>(hidden, Wgk, gkbuf);
    cumsum_kernel<<<NH, 32, 0, stream>>>(gkbuf, Gbuf);
    rope_kernel<<<(TLEN * HDIM / 2) / 256, 256, 0, stream>>>(qf, qbm);
    rope_kernel<<<(TLEN * HDIM / 2) / 256, 256, 0, stream>>>(kf, kbm);
    cast_kernel<<<(TLEN * HDIM) / 256, 256, 0, stream>>>(vf, vbm, TLEN * HDIM);
    attn_kernel<<<dim3(NH, TLEN / 64), 128, 0, stream>>>(qbm, kbm, vbm, Gbuf, gf, obm);
    gemm_bt<bf16><<<gg, 256, 0, stream>>>(obm, Wo, out, TLEN, HDIM, HDIM);
}
